// Kandinsky5NablaAttentionProcessor_25821343383998
// MI455X (gfx1250) — matrix operand order repaired and checked by reading, not compiled, not run on hardware
//
#include <hip/hip_runtime.h>

// Problem constants (match reference)
#define BB 2
#define SS 2048
#define HH 16
#define DD 64
#define BLKSZ 64
#define S1 (SS / BLKSZ)      // 32 block rows/cols
#define HD (HH * DD)         // 1024: stride between consecutive s in [B,S,H,D]
#define KEEP_THR 0.1f        // 1 - 0.9
#define LOG2E 1.44269504088896340736f

typedef __attribute__((ext_vector_type(16))) _Float16 v16h;
typedef __attribute__((ext_vector_type(8)))  float    v8f;

union FragA { v16h v; uint4 q[2]; };

// ---- DPP16 row (16-lane) reductions: pure-VALU, no LDS traffic -------------
// row_ror:n ctrl = 0x120+n ; rotate-combine by 1,2,4,8 -> full 16-lane reduce.
template <int CTRL>
__device__ __forceinline__ float dpp_rot(float x) {
  return __int_as_float(
      __builtin_amdgcn_update_dpp(0, __float_as_int(x), CTRL, 0xf, 0xf, true));
}
__device__ __forceinline__ float rowmax16(float x) {
  x = fmaxf(x, dpp_rot<0x121>(x));
  x = fmaxf(x, dpp_rot<0x122>(x));
  x = fmaxf(x, dpp_rot<0x124>(x));
  x = fmaxf(x, dpp_rot<0x128>(x));
  return x;
}
__device__ __forceinline__ float rowsum16(float x) {
  x += dpp_rot<0x121>(x);
  x += dpp_rot<0x122>(x);
  x += dpp_rot<0x124>(x);
  x += dpp_rot<0x128>(x);
  return x;
}

// ---------------------------------------------------------------------------
// Kernel 1: nabla block mask.  One workgroup per (b,h).
//   qa/ka = block-mean pooled Q/K  -> m = softmax(qa ka^T / 8)
//   keep[j] = (inclusive prefix sum of m in stable-ascending order) >= 0.1
//   mask word (32 bits, bit j = keep block j) per (b,h,block-row).
// ---------------------------------------------------------------------------
__global__ __launch_bounds__(256)
void nabla_mask_kernel(const float* __restrict__ q, const float* __restrict__ k,
                       const unsigned char* __restrict__ sta,
                       unsigned int* __restrict__ mask) {
  __shared__ float qa[S1][DD];
  __shared__ float ka[S1][DD];
  __shared__ float msm[S1][S1 + 1];

  const int bh = blockIdx.x;
  const int b = bh / HH, h = bh % HH;
  const int t = threadIdx.x;

  // Block-mean pooling (coalesced along d).
  for (int o = t; o < S1 * DD; o += 256) {
    const int blk = o >> 6, d = o & 63;
    const size_t base = ((size_t)(b * SS + blk * BLKSZ) * HH + h) * DD + d;
    float sq = 0.f, sk = 0.f;
    for (int r = 0; r < BLKSZ; ++r) {
      sq += q[base + (size_t)r * HD];
      sk += k[base + (size_t)r * HD];
    }
    qa[blk][d] = sq * (1.0f / BLKSZ);
    ka[blk][d] = sk * (1.0f / BLKSZ);
  }
  __syncthreads();

  if (t < S1) {
    const int i = t;                      // block row owned by this lane
    float mx = -3.0e38f;
    for (int j = 0; j < S1; ++j) {
      float dot = 0.f;
      for (int d = 0; d < DD; ++d) dot += qa[i][d] * ka[j][d];
      dot *= 0.125f;                      // 1/sqrt(64)
      msm[i][j] = dot;
      mx = fmaxf(mx, dot);
    }
    float sum = 0.f;
    for (int j = 0; j < S1; ++j) { float e = __expf(msm[i][j] - mx); msm[i][j] = e; sum += e; }
    const float inv = 1.0f / sum;
    for (int j = 0; j < S1; ++j) msm[i][j] *= inv;

    unsigned int bits = 0u;
    for (int j = 0; j < S1; ++j) {
      const float mj = msm[i][j];
      float prefix = 0.f;
      for (int kk = 0; kk < S1; ++kk) {   // stable ascending order: (val, idx)
        const float mk = msm[i][kk];
        if (mk < mj || (mk == mj && kk <= j)) prefix += mk;
      }
      bool keep = (prefix >= KEEP_THR);
      if (sta[i * S1 + j]) keep = true;   // OR with sliding-tile mask
      if (keep) bits |= (1u << j);
    }
    mask[bh * S1 + i] = bits;
  }
}

// ---------------------------------------------------------------------------
// Kernel 2: block-sparse flash attention.
//   grid (s1, H, B); 128 threads = 4 waves; wave w owns query rows [16w,16w+16).
//   LDS: Q (f16, prescaled by softmax_scale*log2e), K (f16), V^T (f16),
//   per-wave P staging.  16x16x32 f16 WMMA for QK^T and PV; streaming softmax
//   runs in the log2 domain (v_exp_f32 directly, no extra multiply).
// ---------------------------------------------------------------------------
#define LSTR 72   // LDS row stride in halves: 144 B, keeps 16-B alignment

__global__ __launch_bounds__(128)
void nabla_attn_kernel(const float* __restrict__ q, const float* __restrict__ k,
                       const float* __restrict__ v,
                       const unsigned int* __restrict__ mask,
                       float* __restrict__ out) {
  __shared__ _Float16 Qh[BLKSZ][LSTR];
  __shared__ _Float16 Kh[BLKSZ][LSTR];
  __shared__ _Float16 Vt[DD][LSTR];          // transposed: [feat][key]
  __shared__ _Float16 Pst[4][16][LSTR];      // per-wave P staging

  const int qblk = blockIdx.x, h = blockIdx.y, b = blockIdx.z;
  const int tid = threadIdx.x;
  const int wave = tid >> 5;
  const int lane = tid & 31;
  const int lr = lane & 15;                  // row/col within a 16-tile
  const int halfsel = lane >> 4;             // lane group 0 / 1
  const int abase = halfsel * 8;             // A-frag K-base (ISA 16-bit A layout)
  const int bbase = halfsel * 8;             // B-frag K-base (same layout as A)
  const int m0 = wave * 16;

  // Stage Q, pre-scaled so scores are already in log2 domain.
  for (int o = tid; o < BLKSZ * DD; o += 128) {
    const int r = o >> 6, d = o & 63;
    const float x = q[((size_t)(b * SS + qblk * BLKSZ + r) * HH + h) * DD + d];
    Qh[r][d] = (_Float16)(x * (0.125f * LOG2E));
  }
  __syncthreads();

  // Preload Q A-fragments (two K-chunks of 32).
  FragA aq[2];
#pragma unroll
  for (int kc = 0; kc < 2; ++kc) {
    const _Float16* p = &Qh[m0 + lr][kc * 32 + abase];
    aq[kc].q[0] = *(const uint4*)p;
    aq[kc].q[1] = *(const uint4*)(p + 16);
  }

  float rm[8], rs[8];
  v8f acc[4];
  {
    v8f z = {};
#pragma unroll
    for (int t = 0; t < 4; ++t) acc[t] = z;
#pragma unroll
    for (int vi = 0; vi < 8; ++vi) { rm[vi] = -1.0e30f; rs[vi] = 0.f; }
  }

  const unsigned int word = mask[(b * HH + h) * S1 + qblk];  // uniform per WG

  for (int j = 0; j < S1; ++j) {
    if (!((word >> j) & 1u)) continue;       // uniform branch -> barriers safe
    __syncthreads();                         // previous-iter LDS reads done

    // Stage K block and V^T block (f32 -> f16).
    for (int o = tid; o < BLKSZ * DD; o += 128) {
      const int r = o >> 6, d = o & 63;
      const size_t gi = ((size_t)(b * SS + j * BLKSZ + r) * HH + h) * DD + d;
      Kh[r][d] = (_Float16)k[gi];
      Vt[d][r] = (_Float16)v[gi];
    }
    // Prefetch next kept block (emits global_prefetch_b8).
    {
      int jn = -1;
      for (int jj = j + 1; jj < S1; ++jj)
        if ((word >> jj) & 1u) { jn = jj; break; }
      if (jn >= 0) {
        const int r = tid & 63;
        const size_t gi = ((size_t)(b * SS + jn * BLKSZ + r) * HH + h) * DD;
        const float* p = (tid < 64) ? (k + gi) : (v + gi);
        __builtin_prefetch(p, 0, 0);
      }
    }
    __syncthreads();

    // S = Q * K^T  (4 N-tiles x 2 K-chunks = 8 WMMAs per wave)
    v8f st[4];
#pragma unroll
    for (int t = 0; t < 4; ++t) {
      v8f c = {};
#pragma unroll
      for (int kc = 0; kc < 2; ++kc) {
        FragA bk;
        const _Float16* p = &Kh[t * 16 + lr][kc * 32 + bbase];
        bk.q[0] = *(const uint4*)p;
        bk.q[1] = *(const uint4*)(p + 16);
        c = __builtin_amdgcn_wmma_f32_16x16x32_f16(
            false, aq[kc].v, false, bk.v, (short)0, c, false, false);
      }
      st[t] = c;
    }

    // Streaming softmax update (log2 domain; DPP row reductions, no LDS).
#pragma unroll
    for (int vi = 0; vi < 8; ++vi) {
      float bmax = fmaxf(fmaxf(st[0][vi], st[1][vi]), fmaxf(st[2][vi], st[3][vi]));
      bmax = rowmax16(bmax);
      const float mnew = fmaxf(rm[vi], bmax);
      const float corr = __builtin_amdgcn_exp2f(rm[vi] - mnew);
      float psum = 0.f;
#pragma unroll
      for (int t = 0; t < 4; ++t) {
        const float p = __builtin_amdgcn_exp2f(st[t][vi] - mnew);
        st[t][vi] = p;
        psum += p;
      }
      psum = rowsum16(psum);
      rs[vi] = rs[vi] * corr + psum;
      rm[vi] = mnew;
      const int row = vi + abase;            // C-layout row for this lane group
#pragma unroll
      for (int t = 0; t < 4; ++t) {
        acc[t][vi] *= corr;
        Pst[wave][row][t * 16 + lr] = (_Float16)st[t][vi];
      }
    }

    // O += P * V  (per-wave P region; same-wave LDS ops are in-order)
#pragma unroll
    for (int kc = 0; kc < 2; ++kc) {
      FragA pa;
      const _Float16* pp = &Pst[wave][lr][kc * 32 + abase];
      pa.q[0] = *(const uint4*)pp;
      pa.q[1] = *(const uint4*)(pp + 16);
#pragma unroll
      for (int t = 0; t < 4; ++t) {
        FragA bv;
        const _Float16* vp = &Vt[t * 16 + lr][kc * 32 + bbase];
        bv.q[0] = *(const uint4*)vp;
        bv.q[1] = *(const uint4*)(vp + 16);
        acc[t] = __builtin_amdgcn_wmma_f32_16x16x32_f16(
            false, pa.v, false, bv.v, (short)0, acc[t], false, false);
      }
    }
  }

  // Epilogue: normalize and store to [B, S, H*D].
  // Each store covers two 64-byte contiguous segments (half-wave per row).
#pragma unroll
  for (int vi = 0; vi < 8; ++vi) {
    const float inv = 1.0f / rs[vi];
    const int srow = qblk * BLKSZ + m0 + vi + abase;
    const size_t ob = ((size_t)b * SS + srow) * HD + (size_t)h * DD;
#pragma unroll
    for (int t = 0; t < 4; ++t) out[ob + t * 16 + lr] = acc[t][vi] * inv;
  }
}

// ---------------------------------------------------------------------------
extern "C" void kernel_launch(void* const* d_in, const int* in_sizes, int n_in,
                              void* d_out, int out_size, void* d_ws, size_t ws_size,
                              hipStream_t stream) {
  const float* q = (const float*)d_in[0];
  const float* k = (const float*)d_in[1];
  const float* v = (const float*)d_in[2];
  const unsigned char* sta = (const unsigned char*)d_in[3];
  unsigned int* mask = (unsigned int*)d_ws;   // B*H*S1 = 1024 words
  float* out = (float*)d_out;

  nabla_mask_kernel<<<dim3(BB * HH), 256, 0, stream>>>(q, k, sta, mask);
  nabla_attn_kernel<<<dim3(S1, HH, BB), 128, 0, stream>>>(q, k, v, mask, out);
}
